// EncapsulationGNN_2061584302396
// MI455X (gfx1250) — matrix operand order repaired and checked by reading, not compiled, not run on hardware
//
#include <hip/hip_runtime.h>
#include <math.h>

// ---------------------------------------------------------------------------
// EncapsulationGNN on MI455X (gfx1250, wave32, WMMA 16x16x32 f16 -> f32)
// ---------------------------------------------------------------------------

typedef __attribute__((ext_vector_type(16))) _Float16 v16h;
typedef __attribute__((ext_vector_type(8)))  _Float16 v8h;
typedef __attribute__((ext_vector_type(4)))  _Float16 v4h;
typedef __attribute__((ext_vector_type(8)))  float    v8f;
typedef __attribute__((ext_vector_type(4)))  float    v4f;

#define H 128
#define NT 8          // N tiles of 16 covering H=128
#define WPB 4         // waves per block
#define KE 288        // 259 padded up to multiple of 32
#define KN 256        // node-update K (128 h + 128 aggr)

union FragU { v16h v; v8h h[2]; };

// A fragment: 16x32 f16 tile from row-major [16][ldk] f16 LDS buffer.
// ISA layout: lanes 0-15 row M=lane hold K {0..7, 16..23}; lanes 16-31 hold
// K {8..15, 24..31} of row M=lane-16.
__device__ __forceinline__ v16h load_a_frag(const _Float16* buf, int ldk,
                                            int k0, int lane) {
    int row = lane & 15;
    int hi  = lane >> 4;
    const _Float16* p = buf + row * ldk + k0 + hi * 8;
    FragU f;
    f.h[0] = *(const v8h*)(p);
    f.h[1] = *(const v8h*)(p + 16);
    return f.v;
}

// B fragment: 32x16 f16 tile, weights stored transposed [N][ldk] f16 in LDS.
// lanes 0-15: column N = ntile*16+lane, K {0..7, 16..23}; lanes 16-31: K {8..15, 24..31}.
__device__ __forceinline__ v16h load_b_frag(const _Float16* wT, int ldk,
                                            int k0, int ntile, int lane) {
    int n  = ntile * 16 + (lane & 15);
    int hi = lane >> 4;
    const _Float16* p = wT + n * ldk + k0 + hi * 8;
    FragU f;
    f.h[0] = *(const v8h*)(p);
    f.h[1] = *(const v8h*)(p + 16);
    return f.v;
}

__device__ __forceinline__ v4h cvt4(v4f x) {
    v4h r;
    r[0] = (_Float16)x[0]; r[1] = (_Float16)x[1];
    r[2] = (_Float16)x[2]; r[3] = (_Float16)x[3];
    return r;
}

__global__ void zero_kernel(float* __restrict__ p, int n) {
    int i = blockIdx.x * blockDim.x + threadIdx.x;
    if (i < n) p[i] = 0.0f;
}

// h0 = concat(x, emb[bead]) @ W_in + b_in      [N,37]@[37,128]
__global__ void embed_kernel(const float* __restrict__ x,
                             const int* __restrict__ bead,
                             const float* __restrict__ emb,
                             const float* __restrict__ Win,
                             const float* __restrict__ bin,
                             float* __restrict__ h, int Nn) {
    int idx = blockIdx.x * blockDim.x + threadIdx.x;
    if (idx >= Nn * H) return;
    int node = idx >> 7;
    int f    = idx & 127;
    float a = bin[f];
    for (int j = 0; j < 5; ++j) a += x[node * 5 + j] * Win[j * H + f];
    const float* e = emb + bead[node] * 32;
    for (int j = 0; j < 32; ++j) a += e[j] * Win[(5 + j) * H + f];
    h[idx] = a;
}

// Per-edge message MLP + scatter-add.  One wave per 16-edge tile.
__global__ void edge_msg_kernel(const float* __restrict__ h,
                                const int* __restrict__ ei,      // [2][E]
                                const float* __restrict__ eattr, // [E][3]
                                const float* __restrict__ W1,    // [259][128]
                                const float* __restrict__ b1,
                                const float* __restrict__ W2,    // [128][128]
                                const float* __restrict__ b2,
                                float* __restrict__ aggr,
                                int E, int Nn) {
    extern __shared__ char smem[];
    _Float16* w1T  = (_Float16*)smem;             // [128][KE]
    _Float16* w2T  = w1T + H * KE;                // [128][128]
    float*    b1s  = (float*)(w2T + H * H);       // 128
    float*    b2s  = b1s + H;                     // 128
    _Float16* aAll = (_Float16*)(b2s + H);        // WPB * 16*KE
    _Float16* hAll = aAll + WPB * 16 * KE;        // WPB * 16*H
    int*      iAll = (int*)(hAll + WPB * 16 * H); // WPB * 48

    int tid = threadIdx.x;
    // stage weights transposed (f32 -> f16); zero K padding region
    for (int i = tid; i < H * KE; i += blockDim.x) w1T[i] = (_Float16)0.0f;
    __syncthreads();
    for (int i = tid; i < 259 * H; i += blockDim.x) {
        int k = i >> 7, n = i & 127;
        w1T[n * KE + k] = (_Float16)W1[i];
    }
    for (int i = tid; i < H * H; i += blockDim.x) {
        int k = i >> 7, n = i & 127;
        w2T[n * H + k] = (_Float16)W2[i];
    }
    for (int i = tid; i < H; i += blockDim.x) { b1s[i] = b1[i]; b2s[i] = b2[i]; }
    __syncthreads();

    int waveId = tid >> 5;
    int lane   = tid & 31;
    int col    = lane & 15;
    int hi     = lane >> 4;
    _Float16* aStage = aAll + waveId * 16 * KE;
    _Float16* hStage = hAll + waveId * 16 * H;
    int* srcS = iAll + waveId * 48;
    int* dstS = srcS + 16;
    int* valS = dstS + 16;

    int gw    = blockIdx.x * WPB + waveId;
    int nw    = gridDim.x * WPB;
    int Etot  = E + Nn;
    int tiles = (Etot + 15) >> 4;

    const v4h z4 = {(_Float16)0, (_Float16)0, (_Float16)0, (_Float16)0};

    for (int tile = gw; tile < tiles; tile += nw) {
        // stage edge indices for this tile (wave-private)
        if (lane < 16) {
            int eid = tile * 16 + lane;
            int s = 0, d = 0, v = 0;
            if (eid < E)         { s = ei[eid]; d = ei[E + eid]; v = 1; }
            else if (eid < Etot) { s = eid - E; d = s;           v = 1; }
            srcS[lane] = s; dstS[lane] = d; valS[lane] = v;
        }
        // gather msg_in rows: [h[src] | h[dst] | ea | 0-pad] -> f16 LDS
        // 32 lanes x float4 covers each 128-float feature row (wide & branch-free)
        for (int r = 0; r < 16; ++r) {
            _Float16* rowp = aStage + r * KE;
            int eid = tile * 16 + r;
            if (valS[r]) {                       // wave-uniform branch
                int s = srcS[r], d = dstS[r];
                v4f hs = *(const v4f*)(h + s * H + lane * 4);
                v4f hd = *(const v4f*)(h + d * H + lane * 4);
                *(v4h*)(rowp + lane * 4)     = cvt4(hs);
                *(v4h*)(rowp + H + lane * 4) = cvt4(hd);
                float ev = (lane < 3 && eid < E) ? eattr[eid * 3 + lane] : 0.0f;
                rowp[2 * H + lane] = (_Float16)ev;
            } else {
                *(v4h*)(rowp + lane * 4)     = z4;
                *(v4h*)(rowp + H + lane * 4) = z4;
                rowp[2 * H + lane] = (_Float16)0.0f;
            }
        }
        // GEMM1: [16,288] x [288,128], bias folded into accumulator init
        v8f c[NT];
        for (int t = 0; t < NT; ++t) {
            float bb = b1s[t * 16 + col];
            c[t] = (v8f){bb, bb, bb, bb, bb, bb, bb, bb};
        }
        for (int k0 = 0; k0 < KE; k0 += 32) {
            v16h a = load_a_frag(aStage, KE, k0, lane);
            for (int t = 0; t < NT; ++t) {
                v16h b = load_b_frag(w1T, KE, k0, t, lane);
                c[t] = __builtin_amdgcn_wmma_f32_16x16x32_f16(
                    false, a, false, b, (short)0, c[t], false, false);
            }
        }
        // relu, re-stage as f16 A operand
        for (int t = 0; t < NT; ++t) {
            int n = t * 16 + col;
            for (int r = 0; r < 8; ++r) {
                int M = r + 8 * hi;
                float v = c[t][r];
                v = v > 0.0f ? v : 0.0f;
                hStage[M * H + n] = (_Float16)v;
            }
        }
        // GEMM2: [16,128] x [128,128], bias folded into accumulator init
        v8f o[NT];
        for (int t = 0; t < NT; ++t) {
            float bb = b2s[t * 16 + col];
            o[t] = (v8f){bb, bb, bb, bb, bb, bb, bb, bb};
        }
        for (int k0 = 0; k0 < H; k0 += 32) {
            v16h a = load_a_frag(hStage, H, k0, lane);
            for (int t = 0; t < NT; ++t) {
                v16h b = load_b_frag(w2T, H, k0, t, lane);
                o[t] = __builtin_amdgcn_wmma_f32_16x16x32_f16(
                    false, a, false, b, (short)0, o[t], false, false);
            }
        }
        // scatter-add m into aggr[dst]
        for (int t = 0; t < NT; ++t) {
            int n = t * 16 + col;
            for (int r = 0; r < 8; ++r) {
                int M = r + 8 * hi;
                if (valS[M]) {
                    atomicAdd(&aggr[dstS[M] * H + n], o[t][r]);
                }
            }
        }
    }
}

// Node update MLP + BN(eval) + ReLU + residual.  One wave per 16-node tile.
__global__ void node_update_kernel(const float* __restrict__ h,
                                   const float* __restrict__ aggr,
                                   const float* __restrict__ W1, // [256][128]
                                   const float* __restrict__ b1,
                                   const float* __restrict__ W2, // [128][128]
                                   const float* __restrict__ b2,
                                   const float* __restrict__ gamma,
                                   const float* __restrict__ beta,
                                   float* __restrict__ hout,
                                   int Nn, int residual) {
    extern __shared__ char smem[];
    _Float16* w1T  = (_Float16*)smem;             // [128][KN]
    _Float16* w2T  = w1T + H * KN;                // [128][128]
    float*    b1s  = (float*)(w2T + H * H);
    float*    b2s  = b1s + H;
    float*    gms  = b2s + H;
    float*    bts  = gms + H;
    _Float16* aAll = (_Float16*)(bts + H);        // WPB * 16*KN
    _Float16* hAll = aAll + WPB * 16 * KN;        // WPB * 16*H

    int tid = threadIdx.x;
    for (int i = tid; i < KN * H; i += blockDim.x) {
        int k = i >> 7, n = i & 127;
        w1T[n * KN + k] = (_Float16)W1[i];
    }
    for (int i = tid; i < H * H; i += blockDim.x) {
        int k = i >> 7, n = i & 127;
        w2T[n * H + k] = (_Float16)W2[i];
    }
    const float bnscale = rsqrtf(1.0f + 1e-5f);
    for (int i = tid; i < H; i += blockDim.x) {
        b1s[i] = b1[i]; b2s[i] = b2[i];
        gms[i] = gamma[i] * bnscale; bts[i] = beta[i];
    }
    __syncthreads();

    int waveId = tid >> 5;
    int lane   = tid & 31;
    int col    = lane & 15;
    int hi     = lane >> 4;
    _Float16* aStage = aAll + waveId * 16 * KN;
    _Float16* hStage = hAll + waveId * 16 * H;

    int gw    = blockIdx.x * WPB + waveId;
    int nw    = gridDim.x * WPB;
    int tiles = (Nn + 15) >> 4;

    const v4h z4 = {(_Float16)0, (_Float16)0, (_Float16)0, (_Float16)0};

    for (int tile = gw; tile < tiles; tile += nw) {
        // gather [h | aggr] -> f16 LDS (32 lanes x float4 per row)
        for (int r = 0; r < 16; ++r) {
            int node = tile * 16 + r;
            _Float16* rowp = aStage + r * KN;
            if (node < Nn) {                     // wave-uniform branch
                v4f hv = *(const v4f*)(h + node * H + lane * 4);
                v4f av = *(const v4f*)(aggr + node * H + lane * 4);
                *(v4h*)(rowp + lane * 4)     = cvt4(hv);
                *(v4h*)(rowp + H + lane * 4) = cvt4(av);
            } else {
                *(v4h*)(rowp + lane * 4)     = z4;
                *(v4h*)(rowp + H + lane * 4) = z4;
            }
        }
        // GEMM1: [16,256] x [256,128], bias in accumulator init
        v8f c[NT];
        for (int t = 0; t < NT; ++t) {
            float bb = b1s[t * 16 + col];
            c[t] = (v8f){bb, bb, bb, bb, bb, bb, bb, bb};
        }
        for (int k0 = 0; k0 < KN; k0 += 32) {
            v16h a = load_a_frag(aStage, KN, k0, lane);
            for (int t = 0; t < NT; ++t) {
                v16h b = load_b_frag(w1T, KN, k0, t, lane);
                c[t] = __builtin_amdgcn_wmma_f32_16x16x32_f16(
                    false, a, false, b, (short)0, c[t], false, false);
            }
        }
        for (int t = 0; t < NT; ++t) {
            int n = t * 16 + col;
            for (int r = 0; r < 8; ++r) {
                int M = r + 8 * hi;
                float v = c[t][r];
                v = v > 0.0f ? v : 0.0f;
                hStage[M * H + n] = (_Float16)v;
            }
        }
        // GEMM2: [16,128] x [128,128], bias in accumulator init
        v8f o[NT];
        for (int t = 0; t < NT; ++t) {
            float bb = b2s[t * 16 + col];
            o[t] = (v8f){bb, bb, bb, bb, bb, bb, bb, bb};
        }
        for (int k0 = 0; k0 < H; k0 += 32) {
            v16h a = load_a_frag(hStage, H, k0, lane);
            for (int t = 0; t < NT; ++t) {
                v16h b = load_b_frag(w2T, H, k0, t, lane);
                o[t] = __builtin_amdgcn_wmma_f32_16x16x32_f16(
                    false, a, false, b, (short)0, o[t], false, false);
            }
        }
        // BN(eval) + ReLU + residual, write h_next
        for (int t = 0; t < NT; ++t) {
            int n = t * 16 + col;
            float gm = gms[n], bt = bts[n];
            for (int r = 0; r < 8; ++r) {
                int M = r + 8 * hi;
                int node = tile * 16 + M;
                if (node < Nn) {
                    float hn = o[t][r] * gm + bt;
                    hn = hn > 0.0f ? hn : 0.0f;
                    float res = residual ? h[node * H + n] : 0.0f;
                    hout[node * H + n] = hn + res;
                }
            }
        }
    }
}

// global_add_pool
__global__ void pool_kernel(const float* __restrict__ h,
                            const int* __restrict__ batch,
                            float* __restrict__ pooled, int Nn) {
    int idx = blockIdx.x * blockDim.x + threadIdx.x;
    if (idx >= Nn * H) return;
    int node = idx >> 7;
    int f    = idx & 127;
    atomicAdd(&pooled[batch[node] * H + f], h[idx]);
}

// graph features + 4-layer head; one thread per graph (G=64)
__global__ void head_kernel(const float* __restrict__ pooled,
                            const float* __restrict__ na, const float* __restrict__ nb,
                            const float* __restrict__ ad, const float* __restrict__ md,
                            const float* __restrict__ gd, const float* __restrict__ tc,
                            const float* __restrict__ cs, const float* __restrict__ ub,
                            const float* __restrict__ gfpW, const float* __restrict__ gfpb,
                            const float* __restrict__ W1, const float* __restrict__ b1,
                            const float* __restrict__ W2, const float* __restrict__ b2,
                            const float* __restrict__ W3, const float* __restrict__ b3,
                            const float* __restrict__ W4, const float* __restrict__ b4,
                            float* __restrict__ out, int G) {
    int g = blockIdx.x * blockDim.x + threadIdx.x;
    if (g >= G) return;
    float gf[8] = { na[g] / 100.0f, nb[g] / 100.0f, ad[g] / 10.0f, md[g] / 10.0f,
                    gd[g],          tc[g] / 10.0f,  cs[g] / 5.0f,  ub[g] / 50.0f };
    float cat[192];
    for (int f = 0; f < H; ++f) cat[f] = pooled[g * H + f];
    for (int j = 0; j < 64; ++j) {
        float a = gfpb[j];
        for (int k = 0; k < 8; ++k) a += gf[k] * gfpW[k * 64 + j];
        cat[H + j] = a;
    }
    float a1[128];
    for (int j = 0; j < 128; ++j) {
        float a = b1[j];
        for (int k = 0; k < 192; ++k) a += cat[k] * W1[k * 128 + j];
        a1[j] = a > 0.0f ? a : 0.0f;
    }
    float a2[64];
    for (int j = 0; j < 64; ++j) {
        float a = b2[j];
        for (int k = 0; k < 128; ++k) a += a1[k] * W2[k * 64 + j];
        a2[j] = a > 0.0f ? a : 0.0f;
    }
    float a3[32];
    for (int j = 0; j < 32; ++j) {
        float a = b3[j];
        for (int k = 0; k < 64; ++k) a += a2[k] * W3[k * 32 + j];
        a3[j] = a > 0.0f ? a : 0.0f;
    }
    float s = b4[0];
    for (int k = 0; k < 32; ++k) s += a3[k] * W4[k];
    out[g] = 1.0f / (1.0f + expf(-s));
}

extern "C" void kernel_launch(void* const* d_in, const int* in_sizes, int n_in,
                              void* d_out, int out_size, void* d_ws, size_t ws_size,
                              hipStream_t stream) {
    const float* x        = (const float*)d_in[0];
    const float* eattr    = (const float*)d_in[1];
    const int*   eindex   = (const int*)d_in[2];
    const int*   batch    = (const int*)d_in[3];
    const int*   bead     = (const int*)d_in[4];
    const float* num_atoms  = (const float*)d_in[5];
    const float* num_bonds  = (const float*)d_in[6];
    const float* avg_deg    = (const float*)d_in[7];
    const float* max_deg    = (const float*)d_in[8];
    const float* density    = (const float*)d_in[9];
    const float* tot_chg    = (const float*)d_in[10];
    const float* chg_std    = (const float*)d_in[11];
    const float* uniq_bt    = (const float*)d_in[12];
    const float* emb      = (const float*)d_in[13];
    const float* W_in     = (const float*)d_in[14];
    const float* b_in     = (const float*)d_in[15];
    const float* msg_W1   = (const float*)d_in[16];
    const float* msg_b1   = (const float*)d_in[17];
    const float* msg_W2   = (const float*)d_in[18];
    const float* msg_b2   = (const float*)d_in[19];
    const float* upd_W1   = (const float*)d_in[20];
    const float* upd_b1   = (const float*)d_in[21];
    const float* upd_W2   = (const float*)d_in[22];
    const float* upd_b2   = (const float*)d_in[23];
    const float* bn_gamma = (const float*)d_in[24];
    const float* bn_beta  = (const float*)d_in[25];
    const float* gfp_W    = (const float*)d_in[26];
    const float* gfp_b    = (const float*)d_in[27];
    const float* hW1 = (const float*)d_in[28];  const float* hb1 = (const float*)d_in[29];
    const float* hW2 = (const float*)d_in[30];  const float* hb2 = (const float*)d_in[31];
    const float* hW3 = (const float*)d_in[32];  const float* hb3 = (const float*)d_in[33];
    const float* hW4 = (const float*)d_in[34];  const float* hb4 = (const float*)d_in[35];

    const int Nn = in_sizes[3];       // nodes
    const int E  = in_sizes[1] / 3;   // edges
    const int G  = in_sizes[5];       // graphs
    const int L  = 3;

    float* ws     = (float*)d_ws;
    float* hA     = ws;
    float* hB     = ws + (size_t)Nn * H;
    float* aggr   = ws + (size_t)2 * Nn * H;
    float* pooled = ws + (size_t)3 * Nn * H;

    // --- input embedding ---
    {
        int total = Nn * H;
        embed_kernel<<<(total + 255) / 256, 256, 0, stream>>>(
            x, bead, emb, W_in, b_in, hA, Nn);
    }

    const size_t edgeLds = (size_t)H * KE * 2 + (size_t)H * H * 2 + 2 * H * 4
                         + (size_t)WPB * 16 * KE * 2 + (size_t)WPB * 16 * H * 2
                         + (size_t)WPB * 48 * 4;
    const size_t nodeLds = (size_t)H * KN * 2 + (size_t)H * H * 2 + 4 * H * 4
                         + (size_t)WPB * 16 * KN * 2 + (size_t)WPB * 16 * H * 2;

    const float* hcur = hA;
    float*       hnext = hB;
    for (int l = 0; l < L; ++l) {
        zero_kernel<<<(Nn * H + 255) / 256, 256, 0, stream>>>(aggr, Nn * H);

        int etiles  = (E + Nn + 15) / 16;
        int eblocks = (etiles + WPB - 1) / WPB;
        if (eblocks > 2048) eblocks = 2048;
        edge_msg_kernel<<<eblocks, WPB * 32, edgeLds, stream>>>(
            hcur, eindex, eattr,
            msg_W1 + (size_t)l * 259 * H, msg_b1 + l * H,
            msg_W2 + (size_t)l * H * H,   msg_b2 + l * H,
            aggr, E, Nn);

        int ntiles  = (Nn + 15) / 16;
        int nblocks = (ntiles + WPB - 1) / WPB;
        if (nblocks > 2048) nblocks = 2048;
        node_update_kernel<<<nblocks, WPB * 32, nodeLds, stream>>>(
            hcur, aggr,
            upd_W1 + (size_t)l * KN * H, upd_b1 + l * H,
            upd_W2 + (size_t)l * H * H,  upd_b2 + l * H,
            bn_gamma + l * H, bn_beta + l * H,
            hnext, Nn, l > 0 ? 1 : 0);

        const float* t = hcur; hcur = hnext; hnext = (float*)t;
    }

    // --- pooling + head ---
    zero_kernel<<<(G * H + 255) / 256, 256, 0, stream>>>(pooled, G * H);
    pool_kernel<<<(Nn * H + 255) / 256, 256, 0, stream>>>(hcur, batch, pooled, Nn);
    head_kernel<<<1, 64, 0, stream>>>(
        pooled, num_atoms, num_bonds, avg_deg, max_deg, density, tot_chg,
        chg_std, uniq_bt, gfp_W, gfp_b, hW1, hb1, hW2, hb2, hW3, hb3, hW4, hb4,
        (float*)d_out, G);
}
